// Attention_56521769615460
// MI455X (gfx1250) — compile-verified, matrix operand order checked by reading, not run on hardware
//
#include <hip/hip_runtime.h>
#include <hip/hip_bf16.h>

// Problem constants (from the reference)
#define B_  2
#define S_  2048
#define D_  1024
#define H_  16
#define HD_ 64

typedef __bf16 bf16;
typedef __bf16 v16bf __attribute__((ext_vector_type(16)));
typedef float  v8f   __attribute__((ext_vector_type(8)));

union ABFrag {
    v16bf v;
    uint4 q[2];
};

// Load a 16x32 bf16 tile (row-major, leading dim ld in elements) into the
// WMMA A/B fragment layout:
//   lane<16 : row = lane,     K = 0..7   (q[0]) and 16..23 (q[1])
//   lane>=16: row = lane-16,  K = 8..15  (q[0]) and 24..31 (q[1])
__device__ __forceinline__ v16bf load_ab(const bf16* __restrict__ base, int ld) {
    const int lane = threadIdx.x & 31;
    const int row  = lane & 15;
    const int ko   = (lane >> 4) * 8;
    ABFrag f;
    const bf16* p = base + (size_t)row * ld + ko;
    f.q[0] = *(const uint4*)(p);
    f.q[1] = *(const uint4*)(p + 16);
    return f.v;
}

__device__ __forceinline__ v8f wmma_bf16(v16bf a, v16bf b, v8f c) {
    return __builtin_amdgcn_wmma_f32_16x16x32_bf16(
        /*neg_a=*/false, a, /*neg_b=*/false, b,
        /*c_mod=*/(short)0, c, /*reuse_a=*/false, /*reuse_b=*/false);
}

// ---------------------------------------------------------------------------
// In-row (16-lane) xor-butterfly reductions via v_permlane16_b32 (pure VALU,
// avoids ds_bpermute LDS traffic). Both 16-lane halves of the wave use the
// same select pattern, which is exactly what the C-fragment row layout needs.
// ---------------------------------------------------------------------------
__device__ __forceinline__ float permlane_f(float x, int sel0, int sel1) {
    int v = __builtin_bit_cast(int, x);
    int p = __builtin_amdgcn_permlane16(v, v, sel0, sel1, true, false);
    return __builtin_bit_cast(float, p);
}
__device__ __forceinline__ float row_max16(float x) {
    x = fmaxf(x, permlane_f(x, 0x67452301, (int)0xEFCDAB89)); // xor 1
    x = fmaxf(x, permlane_f(x, 0x54761032, (int)0xDCFE98BA)); // xor 2
    x = fmaxf(x, permlane_f(x, 0x32107654, (int)0xBA98FEDC)); // xor 4
    x = fmaxf(x, permlane_f(x, (int)0xFEDCBA98, 0x76543210)); // xor 8
    return x;
}
__device__ __forceinline__ float row_sum16(float x) {
    x += permlane_f(x, 0x67452301, (int)0xEFCDAB89);
    x += permlane_f(x, 0x54761032, (int)0xDCFE98BA);
    x += permlane_f(x, 0x32107654, (int)0xBA98FEDC);
    x += permlane_f(x, (int)0xFEDCBA98, 0x76543210);
    return x;
}

// ---------------------------------------------------------------------------
// fp32 -> bf16 conversion
// ---------------------------------------------------------------------------
__global__ void cvt_f32_bf16(const float* __restrict__ in, bf16* __restrict__ out, int n) {
    int i = blockIdx.x * blockDim.x + threadIdx.x;
    if (i < n) out[i] = (bf16)in[i];
}

// ---------------------------------------------------------------------------
// C[M,N] = A[M,K] @ W[N,K]^T  (bf16 inputs, f32 accumulate, OutT output)
// Block: 128 threads = 4 waves as 2(M) x 2(N); block tile 128x128.
// Each wave: 64x64 => acc[4][4] 16x16 tiles; 16 wmma per K-step of 32.
// ---------------------------------------------------------------------------
template <typename OutT>
__global__ void __launch_bounds__(128)
gemm_xwt(const bf16* __restrict__ A, const bf16* __restrict__ W,
         OutT* __restrict__ C, int M, int N, int K) {
    const int lane = threadIdx.x & 31;
    const int wid  = threadIdx.x >> 5;
    const int wm   = wid >> 1;                    // 0..1
    const int wn   = wid & 1;                     // 0..1
    const int m0   = blockIdx.y * 128 + wm * 64;  // wave M origin
    const int n0   = blockIdx.x * 128 + wn * 64;  // wave N origin

    const v8f vzero = {0.f, 0.f, 0.f, 0.f, 0.f, 0.f, 0.f, 0.f};
    v8f acc[4][4];
    for (int i = 0; i < 4; ++i)
        for (int j = 0; j < 4; ++j) acc[i][j] = vzero;

    for (int k = 0; k < K; k += 32) {
        v16bf a[4], b[4];
#pragma unroll
        for (int i = 0; i < 4; ++i)
            a[i] = load_ab(A + (size_t)(m0 + 16 * i) * K + k, K);
#pragma unroll
        for (int j = 0; j < 4; ++j)
            b[j] = load_ab(W + (size_t)(n0 + 16 * j) * K + k, K);
#pragma unroll
        for (int i = 0; i < 4; ++i)
#pragma unroll
            for (int j = 0; j < 4; ++j)
                acc[i][j] = wmma_bf16(a[i], b[j], acc[i][j]);
    }

    const int half = lane >> 4;
    const int col  = lane & 15;
#pragma unroll
    for (int i = 0; i < 4; ++i)
#pragma unroll
        for (int j = 0; j < 4; ++j)
#pragma unroll
            for (int r = 0; r < 8; ++r) {
                int row = m0 + 16 * i + r + 8 * half;
                int cc  = n0 + 16 * j + col;
                C[(size_t)row * N + cc] = (OutT)acc[i][j][r];
            }
}

// ---------------------------------------------------------------------------
// V [B,S,D] (bf16, head-sliced) -> Vt [B,H,HD,S] (bf16) so PV GEMM B-frags
// are contiguous loads.
// ---------------------------------------------------------------------------
__global__ void transpose_v(const bf16* __restrict__ V, bf16* __restrict__ Vt) {
    int idx = blockIdx.x * blockDim.x + threadIdx.x;
    if (idx >= B_ * S_ * D_) return;
    int s  = idx % S_;
    int t  = idx / S_;
    int hd = t % HD_;
    int t2 = t / HD_;
    int h  = t2 % H_;
    int b  = t2 / H_;
    Vt[idx] = V[((size_t)(b * S_ + s)) * D_ + h * HD_ + hd];
}

// ---------------------------------------------------------------------------
// Flash attention: block = 128 threads = 4 waves; each wave owns 16 q-rows.
// kv tiles of 64 (4 score tiles -> one butterfly per row per pass).
// Online softmax with permlane16 row reductions; P restaged C-frag -> A-frag
// through LDS (16x64 bf16 tile per wave).
// ---------------------------------------------------------------------------
__global__ void __launch_bounds__(128)
flash_attn(const bf16* __restrict__ Q, const bf16* __restrict__ Km,
           const bf16* __restrict__ Vt, const float* __restrict__ mask,
           bf16* __restrict__ ctx) {
    __shared__ bf16 Pbuf[4][16 * 64];

    const int lane = threadIdx.x & 31;
    const int wid  = threadIdx.x >> 5;
    const int b    = blockIdx.z;
    const int h    = blockIdx.y;
    const int q0   = blockIdx.x * 64 + wid * 16;           // absolute q row
    const float scale = 0.03125f;                           // 1/sqrt(1024)

    const bf16* Qh  = Q  + ((size_t)b * S_) * D_ + h * HD_;
    const bf16* Kh  = Km + ((size_t)b * S_) * D_ + h * HD_;
    const bf16* Vth = Vt + ((size_t)(b * H_ + h)) * HD_ * S_;

    const v16bf qa0 = load_ab(Qh + (size_t)q0 * D_ + 0,  D_);
    const v16bf qa1 = load_ab(Qh + (size_t)q0 * D_ + 32, D_);

    const v8f vzero = {0.f, 0.f, 0.f, 0.f, 0.f, 0.f, 0.f, 0.f};
    v8f O[4];
    for (int n = 0; n < 4; ++n) O[n] = vzero;
    float m[8], l[8];
    for (int r = 0; r < 8; ++r) { m[r] = -3.0e38f; l[r] = 0.f; }

    const int half = lane >> 4;
    const int col  = lane & 15;

    for (int j0 = 0; j0 < S_; j0 += 64) {
        // scores: four 16x16 tiles covering cols j0 .. j0+63, K = HD = 64
        v8f s[4];
#pragma unroll
        for (int t = 0; t < 4; ++t) {
            v16bf kb0 = load_ab(Kh + (size_t)(j0 + 16 * t) * D_ + 0,  D_);
            v16bf kb1 = load_ab(Kh + (size_t)(j0 + 16 * t) * D_ + 32, D_);
            s[t] = wmma_bf16(qa0, kb0, vzero);
            s[t] = wmma_bf16(qa1, kb1, s[t]);
        }

        float alpha[8];
#pragma unroll
        for (int r = 0; r < 8; ++r) {
            const int row = q0 + r + 8 * half;
            const float* mrow = mask + (size_t)row * S_ + j0 + col;
            // reference: softmax((qk + mask) / sqrt(D))
            float t0 = (s[0][r] + mrow[0])  * scale;
            float t1 = (s[1][r] + mrow[16]) * scale;
            float t2 = (s[2][r] + mrow[32]) * scale;
            float t3 = (s[3][r] + mrow[48]) * scale;

            float mx = row_max16(fmaxf(fmaxf(t0, t1), fmaxf(t2, t3)));
            const float mnew = fmaxf(m[r], mx);

            const float a  = __expf(m[r] - mnew);
            const float e0 = __expf(t0 - mnew);
            const float e1 = __expf(t1 - mnew);
            const float e2 = __expf(t2 - mnew);
            const float e3 = __expf(t3 - mnew);
            const float sum = row_sum16((e0 + e1) + (e2 + e3));

            l[r] = l[r] * a + sum;
            m[r] = mnew;
            alpha[r] = a;

            // stage P (C-frag layout) into LDS as a row-major 16x64 tile
            bf16* pr = &Pbuf[wid][(r + 8 * half) * 64 + col];
            pr[0]  = (bf16)e0;
            pr[16] = (bf16)e1;
            pr[32] = (bf16)e2;
            pr[48] = (bf16)e3;
        }

        // rescale running output
#pragma unroll
        for (int n = 0; n < 4; ++n)
#pragma unroll
            for (int r = 0; r < 8; ++r) O[n][r] *= alpha[r];

        asm volatile("s_wait_dscnt 0" ::: "memory");

        // reload P as two A-fragments (16x32 each) from LDS
        ABFrag pf0, pf1;
        {
            const bf16* pp = &Pbuf[wid][(lane & 15) * 64 + half * 8];
            pf0.q[0] = *(const uint4*)(pp);
            pf0.q[1] = *(const uint4*)(pp + 16);
            pf1.q[0] = *(const uint4*)(pp + 32);
            pf1.q[1] = *(const uint4*)(pp + 48);
        }
        // O[16x64] += P[16x64] @ V[64x64]; Vt rows are hd, contiguous in s
#pragma unroll
        for (int n = 0; n < 4; ++n) {
            v16bf vb0 = load_ab(Vth + (size_t)(n * 16) * S_ + j0,      S_);
            v16bf vb1 = load_ab(Vth + (size_t)(n * 16) * S_ + j0 + 32, S_);
            O[n] = wmma_bf16(pf0.v, vb0, O[n]);
            O[n] = wmma_bf16(pf1.v, vb1, O[n]);
        }
    }

    // normalize and write context back into [B,S,D] head slice (bf16)
#pragma unroll
    for (int n = 0; n < 4; ++n)
#pragma unroll
        for (int r = 0; r < 8; ++r) {
            const int row = q0 + r + 8 * half;
            const int cc  = h * HD_ + n * 16 + col;
            ctx[((size_t)b * S_ + row) * D_ + cc] = (bf16)(O[n][r] / l[r]);
        }
}

// ---------------------------------------------------------------------------
extern "C" void kernel_launch(void* const* d_in, const int* in_sizes, int n_in,
                              void* d_out, int out_size, void* d_ws, size_t ws_size,
                              hipStream_t stream) {
    (void)in_sizes; (void)n_in; (void)out_size; (void)ws_size;

    const float* x    = (const float*)d_in[0];
    const float* mask = (const float*)d_in[1];
    const float* Wq   = (const float*)d_in[2];
    const float* Wk   = (const float*)d_in[3];
    const float* Wv   = (const float*)d_in[4];
    const float* Wo   = (const float*)d_in[5];
    float* out = (float*)d_out;

    const size_t nX = (size_t)B_ * S_ * D_;  // 4M elements
    const size_t nW = (size_t)D_ * D_;       // 1M elements

    bf16* ws  = (bf16*)d_ws;
    bf16* xb  = ws;            // [B,S,D]
    bf16* wqb = xb  + nX;      // [D,D]
    bf16* wkb = wqb + nW;
    bf16* wvb = wkb + nW;
    bf16* wob = wvb + nW;
    bf16* Qb  = wob + nW;      // [B,S,D]
    bf16* Kb  = Qb  + nX;
    bf16* Vb  = Kb  + nX;
    bf16* Vtb = Vb  + nX;      // [B,H,HD,S]
    bf16* ctx = Vtb + nX;      // [B,S,D]

    const int M = B_ * S_;     // 4096
    const int N = D_;          // 1024
    const int K = D_;          // 1024

    // fp32 -> bf16
    cvt_f32_bf16<<<(int)((nX + 255) / 256), 256, 0, stream>>>(x,  xb,  (int)nX);
    cvt_f32_bf16<<<(int)((nW + 255) / 256), 256, 0, stream>>>(Wq, wqb, (int)nW);
    cvt_f32_bf16<<<(int)((nW + 255) / 256), 256, 0, stream>>>(Wk, wkb, (int)nW);
    cvt_f32_bf16<<<(int)((nW + 255) / 256), 256, 0, stream>>>(Wv, wvb, (int)nW);
    cvt_f32_bf16<<<(int)((nW + 255) / 256), 256, 0, stream>>>(Wo, wob, (int)nW);

    // QKV projections (128x128 block tile, 4 waves)
    dim3 ggrid(N / 128, M / 128);
    gemm_xwt<bf16><<<ggrid, 128, 0, stream>>>(xb, wqb, Qb, M, N, K);
    gemm_xwt<bf16><<<ggrid, 128, 0, stream>>>(xb, wkb, Kb, M, N, K);
    gemm_xwt<bf16><<<ggrid, 128, 0, stream>>>(xb, wvb, Vb, M, N, K);

    // V -> Vt per head
    transpose_v<<<(int)((nX + 255) / 256), 256, 0, stream>>>(Vb, Vtb);

    // flash attention: grid (S/64, H, B), 4 waves per block
    flash_attn<<<dim3(S_ / 64, H_, B_), 128, 0, stream>>>(Qb, Kb, Vtb, mask, ctx);

    // output projection, f32 out
    gemm_xwt<float><<<ggrid, 128, 0, stream>>>(ctx, wob, out, M, N, K);
}
